// NATransformerLayer_34840774705572
// MI455X (gfx1250) — matrix operand order repaired and checked by reading, not compiled, not run on hardware
//
#include <hip/hip_runtime.h>
#include <hip/hip_bf16.h>
#include <math.h>

// ---------------------------------------------------------------------------
// Types for CDNA5 WMMA
// ---------------------------------------------------------------------------
typedef __bf16 bf16_t;
typedef __attribute__((ext_vector_type(16))) __bf16 v16bf;
typedef __attribute__((ext_vector_type(8)))  float  v8f;

union Frag16 {            // 32 bytes: one WMMA bf16 A/B fragment per lane
    v16bf v;
    uint4 u[2];
};

// Problem constants
#define NB   4
#define NH   56
#define NW   56
#define NC   256
#define NTOK (NB*NH*NW)       // 12544
#define HEADS 8
#define HD    32              // head dim
#define KS    7
#define MLPH  1024

// ---------------------------------------------------------------------------
// CDNA5 async global->LDS copy (ASYNCcnt-tracked DMA), 16B per lane.
// lds_addr: wave-relative LDS byte address (flat addr truncated to 32 bits).
// ---------------------------------------------------------------------------
__device__ __forceinline__ void async_copy16(unsigned lds_addr, const void* gptr) {
    asm volatile("global_load_async_to_lds_b128 %0, %1, off"
                 :: "v"(lds_addr), "v"(gptr) : "memory");
}
__device__ __forceinline__ void wait_async0() {
    asm volatile("s_wait_asynccnt 0x0" ::: "memory");
}

// ---------------------------------------------------------------------------
// f32 -> bf16 convert (plain)
// ---------------------------------------------------------------------------
__global__ void cvt_bf16_kernel(const float* __restrict__ in,
                                bf16_t* __restrict__ out, int n) {
    int i = blockIdx.x * 256 + threadIdx.x;
    if (i < n) out[i] = (bf16_t)in[i];
}

// f32 [K][N] -> bf16 transposed [N][K] (one-time per launch; amortized over
// all 98 M-blocks, makes the GEMM B-tile a contiguous row copy for async DMA)
__global__ void cvt_bf16_t_kernel(const float* __restrict__ in,
                                  bf16_t* __restrict__ out, int K, int N) {
    int i = blockIdx.x * 256 + threadIdx.x;
    if (i < K * N) {
        int k = i / N, n = i % N;
        out[(size_t)n * K + k] = (bf16_t)in[i];
    }
}

// ---------------------------------------------------------------------------
// LayerNorm over C=256, one wave32 per token, 8 channels per lane.
// ---------------------------------------------------------------------------
__global__ void ln_kernel(const float* __restrict__ x,
                          const float* __restrict__ g,
                          const float* __restrict__ b,
                          bf16_t* __restrict__ y, int ntok) {
    int tok  = (int)((blockIdx.x * blockDim.x + threadIdx.x) >> 5);
    int lane = threadIdx.x & 31;
    if (tok >= ntok) return;

    const float* xp = x + (size_t)tok * NC + lane * 8;
    float4 a0 = *(const float4*)(xp);
    float4 a1 = *(const float4*)(xp + 4);
    float v[8] = {a0.x, a0.y, a0.z, a0.w, a1.x, a1.y, a1.z, a1.w};

    float s = 0.f;
#pragma unroll
    for (int j = 0; j < 8; j++) s += v[j];
#pragma unroll
    for (int o = 16; o > 0; o >>= 1) s += __shfl_xor(s, o, 32);
    float mean = s * (1.f / 256.f);

    float vs = 0.f;
#pragma unroll
    for (int j = 0; j < 8; j++) { float d = v[j] - mean; vs += d * d; }
#pragma unroll
    for (int o = 16; o > 0; o >>= 1) vs += __shfl_xor(vs, o, 32);
    float inv = rsqrtf(vs * (1.f / 256.f) + 1e-5f);

    const float* gp = g + lane * 8;
    const float* bp = b + lane * 8;
    bf16_t ov[8];
#pragma unroll
    for (int j = 0; j < 8; j++)
        ov[j] = (bf16_t)((v[j] - mean) * inv * gp[j] + bp[j]);
    *(uint4*)(y + (size_t)tok * NC + lane * 8) = *(const uint4*)ov;
}

// ---------------------------------------------------------------------------
// WMMA bf16 GEMM:  out[M,N] = A[M,K] @ Bt[N,K]^T + bias (+epilogue)
//   EPI 0: f32 out = acc + bias
//   EPI 1: f32 out = acc + bias + res
//   EPI 2: bf16 out = gelu(acc + bias)        (exact erf GELU)
//
// Block tile M128 x N128, K-step 64, double-buffered LDS filled with
// global_load_async_to_lds_b128. 8 waves: wm in {0,1} (64 rows), wn in
// {0..3} (32 cols); 8 accumulators / 16 WMMAs per wave per K-step.
// ---------------------------------------------------------------------------
template <int EPI>
__global__ __launch_bounds__(256)
void gemm_bf16_kernel(const bf16_t* __restrict__ A, int lda,
                      const bf16_t* __restrict__ Bt, int ldbt,
                      const float* __restrict__ bias,
                      const float* __restrict__ res,
                      void* __restrict__ out, int ldc,
                      int M, int N, int K) {
    // [buf][A:128x64 | B:128x64] bf16, 32KB per buffer
    __shared__ __align__(16) bf16_t sm[2][256 * 64];

    const int t      = threadIdx.x;
    const int lane   = t & 31;
    const int wv     = t >> 5;
    const int wn     = wv & 3;
    const int wm     = wv >> 2;
    const int blockM = blockIdx.x * 128;
    const int blockN = blockIdx.y * 128;

    v8f acc[8];
#pragma unroll
    for (int m = 0; m < 8; m++) acc[m] = (v8f){0.f,0.f,0.f,0.f,0.f,0.f,0.f,0.f};

    const int l16  = lane & 15;
    const int aKof = (lane & 16) ? 8  : 0;   // A fragment K sub-offset
    const int bKof = (lane & 16) ? 8  : 0;   // B fragment K sub-offset

    // issue one 128x64 A tile + 128x64 B tile into buffer `buf` (8 DMAs/thread)
    auto issue_tile = [&](int kt, int buf) {
        bf16_t* Ab = &sm[buf][0];
        bf16_t* Bb = &sm[buf][128 * 64];
#pragma unroll
        for (int i = 0; i < 4; i++) {
            int c   = t + i * 256;        // 0..1023
            int row = c >> 3;
            int seg = (c & 7) * 8;        // element offset, 16B chunks
            async_copy16((unsigned)(size_t)&Ab[row * 64 + seg],
                         A  + (size_t)(blockM + row) * lda  + kt + seg);
            async_copy16((unsigned)(size_t)&Bb[row * 64 + seg],
                         Bt + (size_t)(blockN + row) * ldbt + kt + seg);
        }
    };

    const int ntile = K >> 6;
    issue_tile(0, 0);

    for (int it = 0; it < ntile; ++it) {
        wait_async0();        // this wave's tile-it DMAs have landed in LDS
        __syncthreads();      // all waves' tile-it DMAs visible; prev reads done
        if (it + 1 < ntile) issue_tile((it + 1) << 6, (it + 1) & 1);

        const bf16_t* Ab = &sm[it & 1][0];
        const bf16_t* Bb = &sm[it & 1][128 * 64];

#pragma unroll
        for (int kf = 0; kf < 64; kf += 32) {
            Frag16 bf[2];
#pragma unroll
            for (int n2 = 0; n2 < 2; n2++) {
                int bbase = (wn * 32 + n2 * 16 + l16) * 64 + kf + bKof;
                bf[n2].u[0] = *(const uint4*)&Bb[bbase];
                bf[n2].u[1] = *(const uint4*)&Bb[bbase + 16];
            }
#pragma unroll
            for (int m4 = 0; m4 < 4; m4++) {
                Frag16 af;
                int abase = (wm * 64 + m4 * 16 + l16) * 64 + kf + aKof;
                af.u[0] = *(const uint4*)&Ab[abase];
                af.u[1] = *(const uint4*)&Ab[abase + 16];
#pragma unroll
                for (int n2 = 0; n2 < 2; n2++) {
                    acc[m4 * 2 + n2] = __builtin_amdgcn_wmma_f32_16x16x32_bf16(
                        false, af.v, false, bf[n2].v, (short)0,
                        acc[m4 * 2 + n2], false, false);
                }
            }
        }
    }

    // ---- epilogue ---------------------------------------------------------
#pragma unroll
    for (int n2 = 0; n2 < 2; n2++) {
        const int col = blockN + wn * 32 + n2 * 16 + l16;
        const float bc = bias[col];
#pragma unroll
        for (int m4 = 0; m4 < 4; m4++) {
            int rowbase = blockM + wm * 64 + m4 * 16 + ((lane & 16) ? 8 : 0);
#pragma unroll
            for (int r = 0; r < 8; r++) {
                int row = rowbase + r;
                float vv = acc[m4 * 2 + n2][r] + bc;
                if constexpr (EPI == 1) vv += res[(size_t)row * ldc + col];
                if constexpr (EPI == 2) {
                    vv = 0.5f * vv * (1.f + erff(vv * 0.70710678118654752f));
                    ((bf16_t*)out)[(size_t)row * ldc + col] = (bf16_t)vv;
                } else {
                    ((float*)out)[(size_t)row * ldc + col] = vv;
                }
            }
        }
    }
}

// ---------------------------------------------------------------------------
// Neighborhood attention (7x7, 8 heads, d=32). One block per token, one
// wave32 per head; lane = channel. Wave-wide shfl_xor dot-product reductions.
// qkv layout: [tok][768]  (q: 0..255, k: 256..511, v: 512..767)
// ---------------------------------------------------------------------------
__global__ __launch_bounds__(256)
void nat_kernel(const float* __restrict__ qkv,
                const float* __restrict__ rpb,
                bf16_t* __restrict__ out) {
    const int tok  = blockIdx.x;
    const int head = threadIdx.x >> 5;
    const int lane = threadIdx.x & 31;

    const int bimg = tok / (NH * NW);
    const int rem  = tok % (NH * NW);
    const int py   = rem / NW;
    const int px   = rem % NW;
    const int sh   = min(max(py - KS / 2, 0), NH - KS);
    const int sw   = min(max(px - KS / 2, 0), NW - KS);

    const float scale = 0.17677669529663689f;   // 32^-0.5
    const float qv = qkv[(size_t)tok * 768 + head * HD + lane] * scale;
    const float* rp = rpb + head * (2 * KS - 1) * (2 * KS - 1);

    float logits[KS * KS];
#pragma unroll
    for (int u = 0; u < KS; u++) {
#pragma unroll
        for (int v = 0; v < KS; v++) {
            int ih = sh + u, iw = sw + v;
            int nb = bimg * (NH * NW) + ih * NW + iw;
            float kv = qkv[(size_t)nb * 768 + NC + head * HD + lane];
            float p = qv * kv;
#pragma unroll
            for (int o = 16; o > 0; o >>= 1) p += __shfl_xor(p, o, 32);
            logits[u * KS + v] =
                p + rp[(ih - py + KS - 1) * (2 * KS - 1) + (iw - px + KS - 1)];
        }
    }

    float mx = logits[0];
#pragma unroll
    for (int n = 1; n < KS * KS; n++) mx = fmaxf(mx, logits[n]);
    float se = 0.f;
#pragma unroll
    for (int n = 0; n < KS * KS; n++) { logits[n] = expf(logits[n] - mx); se += logits[n]; }
    const float inv = 1.f / se;

    float acc = 0.f;
#pragma unroll
    for (int u = 0; u < KS; u++) {
#pragma unroll
        for (int v = 0; v < KS; v++) {
            int nb = bimg * (NH * NW) + (sh + u) * NW + (sw + v);
            float vv = qkv[(size_t)nb * 768 + 2 * NC + head * HD + lane];
            acc += logits[u * KS + v] * vv;
        }
    }
    out[(size_t)tok * NC + head * HD + lane] = (bf16_t)(acc * inv);
}

// ---------------------------------------------------------------------------
// Launch
// ---------------------------------------------------------------------------
extern "C" void kernel_launch(void* const* d_in, const int* in_sizes, int n_in,
                              void* d_out, int out_size, void* d_ws, size_t ws_size,
                              hipStream_t stream) {
    const float* x       = (const float*)d_in[0];
    const float* norm1_g = (const float*)d_in[1];
    const float* norm1_b = (const float*)d_in[2];
    const float* qkv_w   = (const float*)d_in[3];
    const float* qkv_b   = (const float*)d_in[4];
    const float* rpb     = (const float*)d_in[5];
    const float* proj_w  = (const float*)d_in[6];
    const float* proj_b  = (const float*)d_in[7];
    const float* norm2_g = (const float*)d_in[8];
    const float* norm2_b = (const float*)d_in[9];
    const float* fc1_w   = (const float*)d_in[10];
    const float* fc1_b   = (const float*)d_in[11];
    const float* fc2_w   = (const float*)d_in[12];
    const float* fc2_b   = (const float*)d_in[13];

    // workspace carve-up (256B aligned slabs)
    char* ws = (char*)d_ws;
    size_t off = 0;
    auto alloc = [&](size_t bytes) {
        void* p = ws + off;
        off = (off + bytes + 255) & ~(size_t)255;
        return p;
    };
    bf16_t* qkvWt  = (bf16_t*)alloc((size_t)NC * 3 * NC * 2);   // [768][256]
    bf16_t* projWt = (bf16_t*)alloc((size_t)NC * NC * 2);       // [256][256]
    bf16_t* fc1Wt  = (bf16_t*)alloc((size_t)NC * MLPH * 2);     // [1024][256]
    bf16_t* fc2Wt  = (bf16_t*)alloc((size_t)MLPH * NC * 2);     // [256][1024]
    bf16_t* y1     = (bf16_t*)alloc((size_t)NTOK * NC * 2);
    float*  qkv    = (float*) alloc((size_t)NTOK * 3 * NC * 4);
    bf16_t* attn   = (bf16_t*)alloc((size_t)NTOK * NC * 2);
    float*  x1     = (float*) alloc((size_t)NTOK * NC * 4);
    bf16_t* y2     = (bf16_t*)alloc((size_t)NTOK * NC * 2);
    bf16_t* hbuf   = (bf16_t*)alloc((size_t)NTOK * MLPH * 2);

    // 1) weight downconvert to bf16, transposed to [N][K] for async staging
    auto cvtT = [&](const float* src, bf16_t* dst, int K, int N) {
        cvt_bf16_t_kernel<<<(K * N + 255) / 256, 256, 0, stream>>>(src, dst, K, N);
    };
    cvtT(qkv_w,  qkvWt,  NC,   3 * NC);
    cvtT(proj_w, projWt, NC,   NC);
    cvtT(fc1_w,  fc1Wt,  NC,   MLPH);
    cvtT(fc2_w,  fc2Wt,  MLPH, NC);

    // 2) LN1: x -> y1 (bf16)
    ln_kernel<<<(NTOK + 7) / 8, 256, 0, stream>>>(x, norm1_g, norm1_b, y1, NTOK);

    // 3) QKV GEMM: y1[12544,256] @ W[256,768] + qkv_b -> qkv (f32)
    gemm_bf16_kernel<0><<<dim3(NTOK / 128, (3 * NC) / 128), 256, 0, stream>>>(
        y1, NC, qkvWt, NC, qkv_b, nullptr, qkv, 3 * NC, NTOK, 3 * NC, NC);

    // 4) neighborhood attention -> attn (bf16)
    nat_kernel<<<NTOK, 256, 0, stream>>>(qkv, rpb, attn);

    // 5) proj GEMM + residual: attn @ W + proj_b + x -> x1 (f32)
    gemm_bf16_kernel<1><<<dim3(NTOK / 128, NC / 128), 256, 0, stream>>>(
        attn, NC, projWt, NC, proj_b, x, x1, NC, NTOK, NC, NC);

    // 6) LN2: x1 -> y2 (bf16)
    ln_kernel<<<(NTOK + 7) / 8, 256, 0, stream>>>(x1, norm2_g, norm2_b, y2, NTOK);

    // 7) FC1 + GELU: y2 @ W + fc1_b -> h (bf16)
    gemm_bf16_kernel<2><<<dim3(NTOK / 128, MLPH / 128), 256, 0, stream>>>(
        y2, NC, fc1Wt, NC, fc1_b, nullptr, hbuf, MLPH, NTOK, MLPH, NC);

    // 8) FC2 + residual: h @ W + fc2_b + x1 -> out (f32)
    gemm_bf16_kernel<1><<<dim3(NTOK / 128, NC / 128), 256, 0, stream>>>(
        hbuf, MLPH, fc2Wt, MLPH, fc2_b, x1, d_out, NC, NTOK, NC, MLPH);

    (void)in_sizes; (void)n_in; (void)out_size; (void)ws_size;
}
